// SwisE_66099546686152
// MI455X (gfx1250) — matrix operand order repaired and checked by reading, not compiled, not run on hardware
//
#include <hip/hip_runtime.h>

#define BATCH   32768
#define NEXP    9          // E
#define RANK    50
#define DIMD    450        // E*RANK
#define XLEN    1350       // 27*50 image
#define NPIX    128        // conv output 8x16
#define TILE    16         // batch rows per block (WMMA M)
#define NOISE_EPS 0.01f
#define EPSF    2.2204460492503131e-16f

typedef __attribute__((ext_vector_type(16))) _Float16 v16h;
typedef __attribute__((ext_vector_type(8)))  float    v8f;

__device__ __forceinline__ float softplusf(float x) {
    return (x > 20.f) ? x : log1pf(expf(x));
}
__device__ __forceinline__ float phif(float z) {
    return 0.5f * (1.f + erff(z * 0.70710678118654752f));
}

__global__ void swise_init(float* ws) {
    if (threadIdx.x < 2 * NEXP) ws[threadIdx.x] = 0.f;
}

__global__ __launch_bounds__(256)
void swise_main(const int* __restrict__ queries,
                const int* __restrict__ these_queries,
                const float* __restrict__ entity,
                const float* __restrict__ rel,
                const float* __restrict__ rel_diag,
                const float* __restrict__ bh,
                const float* __restrict__ bt,
                const float* __restrict__ c_param,
                const float* __restrict__ cnn_w,  const float* __restrict__ cnn_b,
                const float* __restrict__ h2e_w,  const float* __restrict__ h2e_b,
                const float* __restrict__ cnnn_w, const float* __restrict__ cnnn_b,
                const float* __restrict__ h2en_w, const float* __restrict__ h2en_b,
                const float* __restrict__ noise,
                float* __restrict__ y_out,
                float* __restrict__ ws_acc) {
    // Full 27x50 input image per row (head 450 + rel 900), f32
    __shared__ float x_lds[TILE][XLEN];                     // 86.4 KB
    // WMMA operands pre-baked in exact ISA 7.12.2 fragment layouts:
    //   [proj][kb][lane][element]  -> two ds_load_b128 per operand per k-step
    __shared__ alignas(32) _Float16 featA_frag[2][4][32][16];   // 8 KB
    __shared__ alignas(32) _Float16 bfrag[2][4][32][16];        // 8 KB
    __shared__ float    ctile[2][TILE][16];                 // clean / raw_std
    __shared__ float    dist2_s[TILE][NEXP];
    __shared__ int      q_s[TILE][3];                       // q0, q1, t2
    __shared__ float    w_s[2][25];
    __shared__ float    wb_s[2];
    __shared__ float    blkImp[NEXP], blkLoad[NEXP];

    const int tid   = threadIdx.x;
    const int tile0 = blockIdx.x * TILE;

    // ---- Phase A: indices, conv weights, B fragments, accumulators ----
    if (tid < TILE) {
        int b = tile0 + tid;
        q_s[tid][0] = queries[b * 3 + 0];
        q_s[tid][1] = queries[b * 3 + 1];
        q_s[tid][2] = these_queries[b * 3 + 2];
    }
    if (tid >= 64  && tid < 64 + 25) w_s[0][tid - 64] = cnn_w[tid - 64];
    if (tid >= 96  && tid < 96 + 25) w_s[1][tid - 96] = cnnn_w[tid - 96];
    if (tid == 128) wb_s[0] = cnn_b[0];
    if (tid == 129) wb_s[1] = cnnn_b[0];
    if (tid < NEXP) { blkImp[tid] = 0.f; blkLoad[tid] = 0.f; }

    // B fragments: thread tid owns bfrag[w][kb][lane][0..15] (contiguous 32B),
// reading 16 f32 weights in A's table k order (zero-pad columns n = 9..15).
    {
        int w    = tid >> 7;
        int kb   = (tid >> 5) & 3;
        int lane = tid & 31;
        int n    = lane & 15;
        int g    = lane >> 4;
        const float* W = w ? h2en_w : h2e_w;
        #pragma unroll
        for (int e2 = 0; e2 < 16; ++e2) {
            int k = kb * 32 + (e2 & 7) + g * 8 + ((e2 >> 3) << 4);
            bfrag[w][kb][lane][e2] =
                (n < NEXP) ? (_Float16)W[n * NPIX + k] : (_Float16)0.f;
        }
    }
    __syncthreads();

    // prefetch the gathered rows used later from global (global_prefetch_b8)
    if (tid < TILE) {
        __builtin_prefetch(entity   + (size_t)q_s[tid][2] * DIMD, 0, 1);
        __builtin_prefetch(rel_diag + (size_t)q_s[tid][1] * DIMD, 0, 1);
    }

    // ---- Phase B: gather full image rows into LDS (16 threads per row) ----
    {
        int r  = tid >> 4;
        int ln = tid & 15;
        const float* hsrc = entity + (size_t)q_s[r][0] * DIMD;
        const float* rsrc = rel    + (size_t)q_s[r][1] * (2 * DIMD);
        for (int c = ln; c < DIMD; c += 16)        x_lds[r][c]        = hsrc[c];
        for (int c = ln; c < 2 * DIMD; c += 16)    x_lds[r][DIMD + c] = rsrc[c];
    }
    __syncthreads();

    // ---- Phase C1: two 5x5 stride-3 convs, pure LDS reads, output straight
    //      into the WMMA A-fragment layout ----
    for (int cpart = 0; cpart < 2; ++cpart) {
        float wreg[25];
        #pragma unroll
        for (int i = 0; i < 25; ++i) wreg[i] = w_s[cpart][i];
        float bias = wb_s[cpart];
        for (int t = tid; t < TILE * NPIX; t += 256) {
            int r  = t >> 7;
            int o  = t & 127;
            int oy = o >> 4, ox = o & 15;
            int base = (oy * 3) * 50 + ox * 3;
            float acc = bias;
            #pragma unroll
            for (int i = 0; i < 5; ++i)
                #pragma unroll
                for (int j = 0; j < 5; ++j)
                    acc += x_lds[r][base + i * 50 + j] * wreg[i * 5 + j];
            // invert A-matrix 16x32 f16 layout: feature index o -> (kb, lane, e2)
            int kb = o >> 5;
            int ko = o & 31;
            int g  = (ko >> 3) & 1;
            int e2 = (ko & 7) + ((ko >> 4) << 3);
            featA_frag[cpart][kb][g * 16 + r][e2] = (_Float16)acc;
        }
    }

    // ---- Phase C2: Givens rotation + squared distance per (row, expert) ----
    if (tid < TILE * NEXP) {
        int r = tid / NEXP, e = tid - r * NEXP;
        const float* rd  = rel_diag + (size_t)q_s[r][1] * DIMD + e * RANK;
        const float* rhs = entity   + (size_t)q_s[r][2] * DIMD + e * RANK;
        const float* hp  = &x_lds[r][e * RANK];               // head slice
        const float* rp  = &x_lds[r][DIMD + e * (2 * RANK)];  // rel[..., :RANK]
        float d = 0.f;
        #pragma unroll 5
        for (int p = 0; p < RANK / 2; ++p) {
            float rd0 = rd[2 * p], rd1 = rd[2 * p + 1];
            float h0 = hp[2 * p],  h1 = hp[2 * p + 1];
            float nrm = fmaxf(sqrtf(rd0 * rd0 + rd1 * rd1), 1e-15f);
            float c0 = rd0 / nrm, c1 = rd1 / nrm;
            float l0 = c0 * h0 - c1 * h1 + rp[2 * p];
            float l1 = c1 * h0 + c0 * h1 + rp[2 * p + 1];
            float t0 = l0 - rhs[2 * p], t1 = l1 - rhs[2 * p + 1];
            d += t0 * t0 + t1 * t1;
        }
        dist2_s[r][e] = d;
    }
    __syncthreads();

    // ---- Phase D: 16x128x16 projections via v_wmma_f32_16x16x32_f16 ----
    // wave 0 -> clean (h2e_w); wave 1 -> raw_std (h2en_w); fragments are
    // contiguous 32B runs in LDS -> ds_load_b128 pairs, no divergence.
    const int wave = tid >> 5;
    const int lane = tid & 31;
    if (wave < 2) {
        const int g  = lane >> 4;
        const int mn = lane & 15;
        v8f c = {};
        #pragma unroll
        for (int kb = 0; kb < 4; ++kb) {
            v16h a  = *(const v16h*)(&featA_frag[wave][kb][lane][0]);
            v16h bm = *(const v16h*)(&bfrag[wave][kb][lane][0]);
            c = __builtin_amdgcn_wmma_f32_16x16x32_f16(
                    false, a, false, bm, (short)0, c, false, false);
        }
        #pragma unroll
        for (int v = 0; v < 8; ++v)
            ctile[wave][v + 8 * g][mn] = c[v];   // C/D layout: M=v+8g, N=lane&15
    }
    __syncthreads();

    // ---- Phase E: gating, aux-loss partials, score, output ----
    if (tid < TILE) {
        int r = tid, b = tile0 + r;
        float cl[NEXP], sd[NEXP], nz[NEXP];
        #pragma unroll
        for (int e = 0; e < NEXP; ++e) {
            cl[e] = ctile[0][r][e] + h2e_b[e];
            float raw = ctile[1][r][e] + h2en_b[e];
            sd[e] = softplusf(raw) + NOISE_EPS;
            nz[e] = cl[e] + noise[b * NEXP + e] * sd[e];
        }
        // top-3 of the 9 noisy logits (value desc, index asc on ties)
        int i0 = 0, i1 = -1, i2 = -1;
        float l0 = nz[0], l1 = -1e30f, l2 = -1e30f;
        #pragma unroll
        for (int e = 1; e < NEXP; ++e) {
            float v = nz[e];
            if (v > l0)      { l2 = l1; i2 = i1; l1 = l0; i1 = i0; l0 = v; i0 = e; }
            else if (v > l1) { l2 = l1; i2 = i1; l1 = v;  i1 = e; }
            else if (v > l2) { l2 = v;  i2 = e; }
        }
        (void)i2;
        // top-2 softmax gates
        float e1 = expf(l1 - l0);
        float inv = 1.f / (1.f + e1);
        float g0 = inv, g1 = e1 * inv;
        // load probabilities (thr_in = 3rd logit, thr_out = 2nd logit)
        #pragma unroll
        for (int e = 0; e < NEXP; ++e) {
            bool is_in = nz[e] > l2;
            float z = (cl[e] - (is_in ? l2 : l1)) / sd[e];
            atomicAdd(&blkLoad[e], phif(z));
        }
        atomicAdd(&blkImp[i0], g0);
        atomicAdd(&blkImp[i1], g1);
        // score for the two gated experts
        float csp = softplusf(c_param[q_s[r][1]]);
        float base = bh[q_s[r][0]] + bt[q_s[r][2]];
        float s = expf(base - csp * dist2_s[r][i0]) +
                  expf(base - csp * dist2_s[r][i1]);
        y_out[b] = logf((s == 0.f) ? EPSF : s);
    }
    __syncthreads();
    if (tid < NEXP) {
        atomicAdd(&ws_acc[tid], blkImp[tid]);          // importance
        atomicAdd(&ws_acc[NEXP + tid], blkLoad[tid]);  // load
    }
}

__global__ void swise_finalize(const float* __restrict__ ws_acc,
                               float* __restrict__ y_out) {
    if (threadIdx.x == 0) {
        float cv[2];
        #pragma unroll
        for (int p = 0; p < 2; ++p) {
            const float* v = ws_acc + p * NEXP;
            float s = 0.f;
            for (int e = 0; e < NEXP; ++e) s += v[e];
            float mean = s / NEXP;
            float var = 0.f;
            for (int e = 0; e < NEXP; ++e) { float d = v[e] - mean; var += d * d; }
            var /= (NEXP - 1);
            cv[p] = var / (mean * mean + 1e-10f);
        }
        y_out[BATCH] = 0.01f * (cv[0] + cv[1]);
    }
}

extern "C" void kernel_launch(void* const* d_in, const int* in_sizes, int n_in,
                              void* d_out, int out_size, void* d_ws, size_t ws_size,
                              hipStream_t stream) {
    (void)in_sizes; (void)n_in; (void)out_size; (void)ws_size;
    const int*   queries  = (const int*)  d_in[0];
    const int*   these_q  = (const int*)  d_in[1];
    const float* entity   = (const float*)d_in[2];
    const float* rel      = (const float*)d_in[3];
    const float* rel_diag = (const float*)d_in[4];
    const float* bh       = (const float*)d_in[5];
    const float* bt       = (const float*)d_in[6];
    const float* c_param  = (const float*)d_in[7];
    const float* cnn_w    = (const float*)d_in[8];
    const float* cnn_b    = (const float*)d_in[9];
    const float* h2e_w    = (const float*)d_in[10];
    const float* h2e_b    = (const float*)d_in[11];
    const float* cnnn_w   = (const float*)d_in[12];
    const float* cnnn_b   = (const float*)d_in[13];
    const float* h2en_w   = (const float*)d_in[14];
    const float* h2en_b   = (const float*)d_in[15];
    const float* noise    = (const float*)d_in[16];
    float* out = (float*)d_out;
    float* ws  = (float*)d_ws;

    swise_init<<<1, 32, 0, stream>>>(ws);
    swise_main<<<BATCH / TILE, 256, 0, stream>>>(
        queries, these_q, entity, rel, rel_diag, bh, bt, c_param,
        cnn_w, cnn_b, h2e_w, h2e_b, cnnn_w, cnnn_b, h2en_w, h2en_b,
        noise, out, ws);
    swise_finalize<<<1, 32, 0, stream>>>(ws, out);
}
